// MinimalRNNCell_15590731284914
// MI455X (gfx1250) — matrix operand order repaired and checked by reading, not compiled, not run on hardware
//
#include <hip/hip_runtime.h>
#include <stdint.h>

#define Bdim 128
#define Tdim 1024
#define Ddim 512
#define Udim 512
#define LCHUNK 32
#define NCHUNK (Tdim / LCHUNK)

typedef __attribute__((ext_vector_type(16))) __bf16 v16bf;
typedef __attribute__((ext_vector_type(8)))  float v8f;

union Frag16 {
    v16bf v;
    uint4 q[2];
};

__device__ __forceinline__ unsigned short f32_to_bf16_bits(float x) {
    union { float f; unsigned u; } a; a.f = x;
    return (unsigned short)(a.u >> 16);
}
__device__ __forceinline__ float bf16_bits_to_f32(unsigned short h) {
    union { unsigned u; float f; } a; a.u = ((unsigned)h) << 16;
    return a.f;
}
// Split fp32 into bf16 hi + bf16 lo residual (a ~= hi + lo, ~16-bit mantissa)
__device__ __forceinline__ void split_bf16(float x, unsigned short& hi, unsigned short& lo) {
    unsigned short h = f32_to_bf16_bits(x);
    float r = x - bf16_bits_to_f32(h);
    hi = h;
    lo = f32_to_bf16_bits(r);
}
// Pack two fp32 into one dword of bf16-hi and one dword of bf16-lo
__device__ __forceinline__ void split_pack2(float x0, float x1,
                                            unsigned& hpack, unsigned& lpack) {
    unsigned short h0, l0, h1, l1;
    split_bf16(x0, h0, l0);
    split_bf16(x1, h1, l1);
    hpack = (unsigned)h0 | ((unsigned)h1 << 16);
    lpack = (unsigned)l0 | ((unsigned)l1 << 16);
}

__device__ __forceinline__ v8f wmma_bf16(v16bf a, v16bf b, v8f c) {
    return __builtin_amdgcn_wmma_f32_16x16x32_bf16(false, a, false, b, (short)0, c, false, false);
}
__device__ __forceinline__ Frag16 loadB(const unsigned short* base, int qi) {
    Frag16 f;
    f.q[0] = ((const uint4*)base)[qi];
    f.q[1] = ((const uint4*)base)[qi + 1];
    return f;
}
__device__ __forceinline__ Frag16 loadA(const unsigned short* arr, int off) {
    Frag16 f;
    f.q[0] = *(const uint4*)(arr + off);
    f.q[1] = *(const uint4*)(arr + off + 8);
    return f;
}

// ---------------------------------------------------------------------------
// Pack kernel + recurrent_kernel (fp32 row-major [K][N]) into bf16 hi/lo in
// the contiguous-k B-fragment layout this code uses (matching its A loads) with v_wmma_f32_16x16x32_bf16:
//   B tile 32(K) x 16(N): lanes 0-15: n=lane,    K=0..15 (elements 0..15)
//                         lanes 16-31: n=lane-16, K=16..31
// Fragment storage: frag = n_tile*16 + kb; ushort offset = frag*512+lane*16+e
// ws layout (ushort): [Whi | Wlo | Rhi | Rlo], each 262144.
// ---------------------------------------------------------------------------
__global__ void rnn_prep_weights(const float* __restrict__ kern,
                                 const float* __restrict__ rker,
                                 unsigned short* __restrict__ ws) {
    int tid = blockIdx.x * blockDim.x + threadIdx.x;
    if (tid >= 2 * Ddim * Udim) return;
    int mat    = tid >> 18;
    int rem    = tid & ((1 << 18) - 1);
    int frag   = rem >> 9;
    int within = rem & 511;
    int lane   = within >> 4;
    int e      = within & 15;
    int n_tile = frag >> 4;
    int kb     = frag & 15;
    int koff   = ((lane >> 4) << 4) + e;
    int k      = kb * 32 + koff;
    int n      = n_tile * 16 + (lane & 15);
    const float* src = (mat == 0) ? kern : rker;
    float w = src[k * Udim + n];
    unsigned short hi, lo;
    split_bf16(w, hi, lo);
    ws[mat * 524288 + rem]          = hi;
    ws[mat * 524288 + 262144 + rem] = lo;
}

// Pack a single fp32 512x512 matrix into bf16 hi/lo B-fragment layout.
__global__ void rnn_pack_single(const float* __restrict__ src,
                                unsigned short* __restrict__ dst) {
    int tid = blockIdx.x * blockDim.x + threadIdx.x;
    if (tid >= Ddim * Udim) return;
    int rem    = tid;
    int frag   = rem >> 9;
    int within = rem & 511;
    int lane   = within >> 4;
    int e      = within & 15;
    int n_tile = frag >> 4;
    int kb     = frag & 15;
    int koff   = ((lane >> 4) << 4) + e;
    int k      = kb * 32 + koff;
    int n      = n_tile * 16 + (lane & 15);
    float w = src[k * Udim + n];
    unsigned short hi, lo;
    split_bf16(w, hi, lo);
    dst[rem]          = hi;
    dst[262144 + rem] = lo;
}

// Naive fp32 matrix squaring dst = src @ src (512x512). Only used 5x on a
// 0.27 GFLOP problem to build R^32 at full fp32 accuracy.
__global__ void rnn_mat_square(const float* __restrict__ src,
                               float* __restrict__ dst) {
    int idx = blockIdx.x * blockDim.x + threadIdx.x;
    if (idx >= Ddim * Udim) return;
    int i = idx >> 9;
    int j = idx & 511;
    float s = 0.f;
    for (int k = 0; k < Udim; ++k)
        s += src[i * Udim + k] * src[k * Udim + j];
    dst[idx] = s;
}

// ---------------------------------------------------------------------------
// Pass A: chunk-local fused scan. Grid (8 batch-tiles, NC chunks) x 512 thr.
// Chunk 0 starts from h0; other chunks start from zero. Writes y_t to out
// and the chunk's last state to ylast (if non-null). Also serves as the
// full sequential fallback (grid (8,1), Lsteps = Tdim, ylast = nullptr).
// ---------------------------------------------------------------------------
__global__ __launch_bounds__(512, 1)
void rnn_scan_chunk_kernel(const float* __restrict__ x,
                           const float* __restrict__ h0,
                           const unsigned short* __restrict__ wpk,
                           float* __restrict__ out,
                           float* __restrict__ ylast,
                           int Lsteps) {
    __shared__ unsigned short sXhi[16 * 512];
    __shared__ unsigned short sXlo[16 * 512];
    __shared__ unsigned short sHhi[16 * 512];
    __shared__ unsigned short sHlo[16 * 512];

    const unsigned short* Whi = wpk;
    const unsigned short* Wlo = wpk + 262144;
    const unsigned short* Rhi = wpk + 524288;
    const unsigned short* Rlo = wpk + 786432;

    const int b0    = blockIdx.x * 16;
    const int chunk = blockIdx.y;
    const int t0    = chunk * Lsteps;
    const int lane  = threadIdx.x & 31;
    const int wave  = threadIdx.x >> 5;
    const int m     = lane & 15;
    const int kgrp  = lane >> 4;
    const int kbase = kgrp * 16;
    const int srow  = threadIdx.x >> 5;
    const int scol  = (threadIdx.x & 31) * 16;

    // ---- init h: h0 for chunk 0, zero otherwise ----
    if (chunk == 0) {
        const float* hp = h0 + (size_t)(b0 + srow) * Udim + scol;
        float4 buf[4];
        buf[0] = *(const float4*)(hp + 0);
        buf[1] = *(const float4*)(hp + 4);
        buf[2] = *(const float4*)(hp + 8);
        buf[3] = *(const float4*)(hp + 12);
        const float* pv = (const float*)buf;
        for (int j = 0; j < 16; j += 2) {
            unsigned hp2, lp2;
            split_pack2(pv[j], pv[j + 1], hp2, lp2);
            *(unsigned*)&sHhi[srow * 512 + scol + j] = hp2;
            *(unsigned*)&sHlo[srow * 512 + scol + j] = lp2;
        }
    } else {
        for (int j = 0; j < 16; j += 2) {
            *(unsigned*)&sHhi[srow * 512 + scol + j] = 0u;
            *(unsigned*)&sHlo[srow * 512 + scol + j] = 0u;
        }
    }
    __syncthreads();

    const int n0a   = wave * 32;
    const int n0b   = wave * 32 + 16;
    const int fbasA = (wave * 2) * 16;
    const int fbasB = (wave * 2 + 1) * 16;

#pragma unroll 1
    for (int i = 0; i < Lsteps; ++i) {
        const int t = t0 + i;
        // ---- phase 1: stage x_t (hi/lo bf16) ----
        {
            const float* xp = x + ((size_t)(b0 + srow) * Tdim + t) * Ddim + scol;
            float4 buf[4];
            buf[0] = *(const float4*)(xp + 0);
            buf[1] = *(const float4*)(xp + 4);
            buf[2] = *(const float4*)(xp + 8);
            buf[3] = *(const float4*)(xp + 12);
            const float* pv = (const float*)buf;
            for (int j = 0; j < 16; j += 2) {
                unsigned hp2, lp2;
                split_pack2(pv[j], pv[j + 1], hp2, lp2);
                *(unsigned*)&sXhi[srow * 512 + scol + j] = hp2;
                *(unsigned*)&sXlo[srow * 512 + scol + j] = lp2;
            }
        }
        __syncthreads();

        // ---- phase 2: acc = x_t @ W + h @ R (hi/lo 3-product WMMA) ----
        v8f acc0 = {};
        v8f acc1 = {};
#pragma unroll 2
        for (int kb = 0; kb < 16; ++kb) {
            const int offA = m * 512 + kb * 32 + kbase;
            Frag16 axh = loadA(sXhi, offA);
            Frag16 axl = loadA(sXlo, offA);
            Frag16 ahh = loadA(sHhi, offA);
            Frag16 ahl = loadA(sHlo, offA);
            {
                const int qi = (fbasA + kb) * 64 + lane * 2;
                Frag16 bwh = loadB(Whi, qi), bwl = loadB(Wlo, qi);
                Frag16 brh = loadB(Rhi, qi), brl = loadB(Rlo, qi);
                acc0 = wmma_bf16(axh.v, bwh.v, acc0);
                acc0 = wmma_bf16(axh.v, bwl.v, acc0);
                acc0 = wmma_bf16(axl.v, bwh.v, acc0);
                acc0 = wmma_bf16(ahh.v, brh.v, acc0);
                acc0 = wmma_bf16(ahh.v, brl.v, acc0);
                acc0 = wmma_bf16(ahl.v, brh.v, acc0);
            }
            {
                const int qi = (fbasB + kb) * 64 + lane * 2;
                Frag16 bwh = loadB(Whi, qi), bwl = loadB(Wlo, qi);
                Frag16 brh = loadB(Rhi, qi), brl = loadB(Rlo, qi);
                acc1 = wmma_bf16(axh.v, bwh.v, acc1);
                acc1 = wmma_bf16(axh.v, bwl.v, acc1);
                acc1 = wmma_bf16(axl.v, bwh.v, acc1);
                acc1 = wmma_bf16(ahh.v, brh.v, acc1);
                acc1 = wmma_bf16(ahh.v, brl.v, acc1);
                acc1 = wmma_bf16(ahl.v, brh.v, acc1);
            }
        }
        __syncthreads();

        // ---- phase 3: commit y_t ----
        for (int r = 0; r < 8; ++r) {
            const int row = r + 8 * kgrp;
            const size_t ob = ((size_t)(b0 + row) * Tdim + t) * (size_t)Udim;
            float va = acc0[r];
            float vb = acc1[r];
            out[ob + n0a + m] = va;
            out[ob + n0b + m] = vb;
            unsigned short hi, lo;
            split_bf16(va, hi, lo);
            sHhi[row * 512 + n0a + m] = hi;
            sHlo[row * 512 + n0a + m] = lo;
            split_bf16(vb, hi, lo);
            sHhi[row * 512 + n0b + m] = hi;
            sHlo[row * 512 + n0b + m] = lo;
            if (i == Lsteps - 1 && ylast != nullptr) {
                const size_t yb = ((size_t)chunk * Bdim + b0 + row) * (size_t)Udim;
                ylast[yb + n0a + m] = va;
                ylast[yb + n0b + m] = vb;
            }
        }
        __syncthreads();
    }
}

// ---------------------------------------------------------------------------
// Pass B: boundary fix-up (sequential over chunks, 8 WGs).
//   carry_c = y_last(c-1) + carry_{c-1} @ R^L,  carry_0 = 0.
// ---------------------------------------------------------------------------
__global__ __launch_bounds__(512, 1)
void rnn_boundary_kernel(const float* __restrict__ ylast,
                         float* __restrict__ carry,
                         const unsigned short* __restrict__ rlpk,
                         int nchunks) {
    __shared__ unsigned short sChi[16 * 512];
    __shared__ unsigned short sClo[16 * 512];

    const unsigned short* RLhi = rlpk;
    const unsigned short* RLlo = rlpk + 262144;

    const int b0    = blockIdx.x * 16;
    const int lane  = threadIdx.x & 31;
    const int wave  = threadIdx.x >> 5;
    const int m     = lane & 15;
    const int kgrp  = lane >> 4;
    const int kbase = kgrp * 16;
    const int srow  = threadIdx.x >> 5;
    const int scol  = (threadIdx.x & 31) * 16;

    for (int j = 0; j < 16; j += 2) {
        *(unsigned*)&sChi[srow * 512 + scol + j] = 0u;
        *(unsigned*)&sClo[srow * 512 + scol + j] = 0u;
    }
    __syncthreads();

    const int n0a   = wave * 32;
    const int n0b   = wave * 32 + 16;
    const int fbasA = (wave * 2) * 16;
    const int fbasB = (wave * 2 + 1) * 16;

#pragma unroll 1
    for (int c = 1; c < nchunks; ++c) {
        v8f acc0 = {};
        v8f acc1 = {};
        // C-init: acc = y_last(c-1)
        for (int r = 0; r < 8; ++r) {
            const int row = r + 8 * kgrp;
            const size_t yb = ((size_t)(c - 1) * Bdim + b0 + row) * (size_t)Udim;
            acc0[r] = ylast[yb + n0a + m];
            acc1[r] = ylast[yb + n0b + m];
        }
        // acc += carry_{c-1} @ R^L
#pragma unroll 2
        for (int kb = 0; kb < 16; ++kb) {
            const int offA = m * 512 + kb * 32 + kbase;
            Frag16 ahh = loadA(sChi, offA);
            Frag16 ahl = loadA(sClo, offA);
            {
                const int qi = (fbasA + kb) * 64 + lane * 2;
                Frag16 brh = loadB(RLhi, qi), brl = loadB(RLlo, qi);
                acc0 = wmma_bf16(ahh.v, brh.v, acc0);
                acc0 = wmma_bf16(ahh.v, brl.v, acc0);
                acc0 = wmma_bf16(ahl.v, brh.v, acc0);
            }
            {
                const int qi = (fbasB + kb) * 64 + lane * 2;
                Frag16 brh = loadB(RLhi, qi), brl = loadB(RLlo, qi);
                acc1 = wmma_bf16(ahh.v, brh.v, acc1);
                acc1 = wmma_bf16(ahh.v, brl.v, acc1);
                acc1 = wmma_bf16(ahl.v, brh.v, acc1);
            }
        }
        __syncthreads();
        for (int r = 0; r < 8; ++r) {
            const int row = r + 8 * kgrp;
            const size_t cb = ((size_t)c * Bdim + b0 + row) * (size_t)Udim;
            float va = acc0[r];
            float vb = acc1[r];
            carry[cb + n0a + m] = va;
            carry[cb + n0b + m] = vb;
            unsigned short hi, lo;
            split_bf16(va, hi, lo);
            sChi[row * 512 + n0a + m] = hi;
            sClo[row * 512 + n0a + m] = lo;
            split_bf16(vb, hi, lo);
            sChi[row * 512 + n0b + m] = hi;
            sClo[row * 512 + n0b + m] = lo;
        }
        __syncthreads();
    }
}

// ---------------------------------------------------------------------------
// Pass C: carry propagation. Grid (8, NC-1); chunk = blockIdx.y + 1.
//   cc = carry_c;  for i in 0..L-1: cc = cc @ R; out[t0+i] += cc
// ---------------------------------------------------------------------------
__global__ __launch_bounds__(512, 1)
void rnn_carry_apply_kernel(const float* __restrict__ carry,
                            const unsigned short* __restrict__ wpk,
                            float* __restrict__ out,
                            int Lsteps) {
    __shared__ unsigned short sChi[16 * 512];
    __shared__ unsigned short sClo[16 * 512];

    const unsigned short* Rhi = wpk + 524288;
    const unsigned short* Rlo = wpk + 786432;

    const int b0    = blockIdx.x * 16;
    const int chunk = blockIdx.y + 1;
    const int t0    = chunk * Lsteps;
    const int lane  = threadIdx.x & 31;
    const int wave  = threadIdx.x >> 5;
    const int m     = lane & 15;
    const int kgrp  = lane >> 4;
    const int kbase = kgrp * 16;
    const int srow  = threadIdx.x >> 5;
    const int scol  = (threadIdx.x & 31) * 16;

    // stage carry_c into LDS as hi/lo bf16
    {
        const float* cp = carry + ((size_t)chunk * Bdim + b0 + srow) * (size_t)Udim + scol;
        float4 buf[4];
        buf[0] = *(const float4*)(cp + 0);
        buf[1] = *(const float4*)(cp + 4);
        buf[2] = *(const float4*)(cp + 8);
        buf[3] = *(const float4*)(cp + 12);
        const float* pv = (const float*)buf;
        for (int j = 0; j < 16; j += 2) {
            unsigned hp2, lp2;
            split_pack2(pv[j], pv[j + 1], hp2, lp2);
            *(unsigned*)&sChi[srow * 512 + scol + j] = hp2;
            *(unsigned*)&sClo[srow * 512 + scol + j] = lp2;
        }
    }
    __syncthreads();

    const int n0a   = wave * 32;
    const int n0b   = wave * 32 + 16;
    const int fbasA = (wave * 2) * 16;
    const int fbasB = (wave * 2 + 1) * 16;

#pragma unroll 1
    for (int i = 0; i < Lsteps; ++i) {
        const int t = t0 + i;
        v8f acc0 = {};
        v8f acc1 = {};
#pragma unroll 2
        for (int kb = 0; kb < 16; ++kb) {
            const int offA = m * 512 + kb * 32 + kbase;
            Frag16 ahh = loadA(sChi, offA);
            Frag16 ahl = loadA(sClo, offA);
            {
                const int qi = (fbasA + kb) * 64 + lane * 2;
                Frag16 brh = loadB(Rhi, qi), brl = loadB(Rlo, qi);
                acc0 = wmma_bf16(ahh.v, brh.v, acc0);
                acc0 = wmma_bf16(ahh.v, brl.v, acc0);
                acc0 = wmma_bf16(ahl.v, brh.v, acc0);
            }
            {
                const int qi = (fbasB + kb) * 64 + lane * 2;
                Frag16 brh = loadB(Rhi, qi), brl = loadB(Rlo, qi);
                acc1 = wmma_bf16(ahh.v, brh.v, acc1);
                acc1 = wmma_bf16(ahh.v, brl.v, acc1);
                acc1 = wmma_bf16(ahl.v, brh.v, acc1);
            }
        }
        __syncthreads();
        for (int r = 0; r < 8; ++r) {
            const int row = r + 8 * kgrp;
            const size_t ob = ((size_t)(b0 + row) * Tdim + t) * (size_t)Udim;
            float va = acc0[r];
            float vb = acc1[r];
            out[ob + n0a + m] += va;   // y_t + carry correction
            out[ob + n0b + m] += vb;
            unsigned short hi, lo;
            split_bf16(va, hi, lo);
            sChi[row * 512 + n0a + m] = hi;
            sClo[row * 512 + n0a + m] = lo;
            split_bf16(vb, hi, lo);
            sChi[row * 512 + n0b + m] = hi;
            sClo[row * 512 + n0b + m] = lo;
        }
        __syncthreads();
    }
}

extern "C" void kernel_launch(void* const* d_in, const int* in_sizes, int n_in,
                              void* d_out, int out_size, void* d_ws, size_t ws_size,
                              hipStream_t stream) {
    const float* x    = (const float*)d_in[0];   // [B, T, D]
    const float* h0   = (const float*)d_in[1];   // [B, U]
    const float* kern = (const float*)d_in[2];   // [D, U]
    const float* rker = (const float*)d_in[3];   // [U, U]
    float* out = (float*)d_out;                  // [B, T, U]

    char* wsb = (char*)d_ws;
    unsigned short* wpk  = (unsigned short*)(wsb);                 // 2 MB: W/R hi+lo packed
    unsigned short* rlpk = (unsigned short*)(wsb + (2u << 20));    // 1 MB: R^L hi+lo packed
    float* powA  = (float*)(wsb + (3u << 20));                     // 1 MB
    float* powB  = (float*)(wsb + (4u << 20));                     // 1 MB
    float* ylast = (float*)(wsb + (5u << 20));                     // 8 MB [NC][B][U]
    float* carry = (float*)(wsb + (13u << 20));                    // 8 MB [NC][B][U]
    const size_t NEED = 21ull << 20;

    // Pack W and R (bf16 hi/lo, WMMA B-fragment layout)
    rnn_prep_weights<<<(2 * Ddim * Udim + 255) / 256, 256, 0, stream>>>(kern, rker, wpk);

    if (ws_size >= NEED) {
        // R^32 via 5 fp32 squarings, then pack
        rnn_mat_square<<<(Ddim * Udim + 255) / 256, 256, 0, stream>>>(rker, powA);  // R^2
        rnn_mat_square<<<(Ddim * Udim + 255) / 256, 256, 0, stream>>>(powA, powB);  // R^4
        rnn_mat_square<<<(Ddim * Udim + 255) / 256, 256, 0, stream>>>(powB, powA);  // R^8
        rnn_mat_square<<<(Ddim * Udim + 255) / 256, 256, 0, stream>>>(powA, powB);  // R^16
        rnn_mat_square<<<(Ddim * Udim + 255) / 256, 256, 0, stream>>>(powB, powA);  // R^32
        rnn_pack_single<<<(Ddim * Udim + 255) / 256, 256, 0, stream>>>(powA, rlpk);

        // Pass A: 256 WGs, depth 32
        rnn_scan_chunk_kernel<<<dim3(Bdim / 16, NCHUNK), 512, 0, stream>>>(
            x, h0, wpk, out, ylast, LCHUNK);
        // Pass B: 8 WGs, depth 31
        rnn_boundary_kernel<<<dim3(Bdim / 16, 1), 512, 0, stream>>>(
            ylast, carry, rlpk, NCHUNK);
        // Pass C: 248 WGs, depth 32
        rnn_carry_apply_kernel<<<dim3(Bdim / 16, NCHUNK - 1), 512, 0, stream>>>(
            carry, wpk, out, LCHUNK);
    } else {
        // Fallback: fully sequential fused scan (needs only the 2 MB pack)
        rnn_scan_chunk_kernel<<<dim3(Bdim / 16, 1), 512, 0, stream>>>(
            x, h0, wpk, out, nullptr, Tdim);
    }
}
